// FlaxQwen2MoeBlocKSparesTop2MLPCollection_35553739276639
// MI455X (gfx1250) — matrix operand order repaired and checked by reading, not compiled, not run on hardware
//
#include <hip/hip_runtime.h>
#include <hip/hip_bf16.h>

// ---------------------------------------------------------------------------
// MoE top-2 SwiGLU MLP for MI455X (gfx1250), bf16 WMMA path.
//   B=4 S=2048 H=2048 I=1408, 8 experts, top-2.
// Only selected token-expert pairs are computed (283 GFLOP instead of 1.13T).
// Pipeline: route -> compact -> gate/up GEMM + SwiGLU (bf16 act to ws)
//           -> down GEMM with atomic scatter-add into d_out.
// LDS layout is K-contiguous for BOTH operands so WMMA fragments load as
// ds_load_b128.  Wave tile 64x32 (mf=4, nf=2): 16 ds_load_b128 per 16 WMMA
// in the gate/up kernel.
// ---------------------------------------------------------------------------

#define NUM_EXPERTS 8
#define TOPK        2
#define B_          4
#define S_          2048
#define H_          2048
#define I_          1408
#define NTOK        (B_ * S_)          // 8192 tokens
#define NPAIR       (NTOK * TOPK)      // 16384 pairs max
#define TILE_M      256
#define MAXSLOTS    (NPAIR + NUM_EXPERTS * TILE_M)   // 18432 padded slots
#define MAXTILES    (MAXSLOTS / TILE_M)              // 72 m-tiles
#define CAP         NPAIR                            // per-expert staging cap
#define LDK         40                               // LDS row stride (u16)

typedef unsigned short u16;
typedef unsigned int   u32;
typedef __attribute__((ext_vector_type(16))) __bf16 v16bf;
typedef __attribute__((ext_vector_type(8)))  float  v8f;

union Frag { uint4 q[2]; v16bf v; };   // 32 bytes = 16 bf16

__device__ __forceinline__ u16 f2bf(float f) {
  unsigned u = __builtin_bit_cast(unsigned, f);
  unsigned r = 0x7FFFu + ((u >> 16) & 1u);   // round-to-nearest-even
  return (u16)((u + r) >> 16);
}
__device__ __forceinline__ u32 pk2(float a, float b) {
  return (u32)f2bf(a) | ((u32)f2bf(b) << 16);
}
__device__ __forceinline__ uint4 pk8(float4 a, float4 b) {
  uint4 r;
  r.x = pk2(a.x, a.y); r.y = pk2(a.z, a.w);
  r.z = pk2(b.x, b.y); r.w = pk2(b.z, b.w);
  return r;
}

// ---------------------------------------------------------------------------
// Kernel 1: routing — bucket (token, weight) pairs per expert.
// ---------------------------------------------------------------------------
__global__ void route_kernel(const int* __restrict__ sel,
                             const float* __restrict__ rw,
                             int* __restrict__ cnt,
                             int* __restrict__ stage_tok,
                             float* __restrict__ stage_w) {
  int t = blockIdx.x * blockDim.x + threadIdx.x;
  if (t >= NTOK) return;
  int   e0 = sel[t * TOPK + 0];
  int   e1 = sel[t * TOPK + 1];
  float w0 = rw[t * TOPK + 0];
  float w1 = rw[t * TOPK + 1];
  if (e1 == e0) w0 += w1;   // identical top-2 picks merge linearly
  int p = atomicAdd(&cnt[e0], 1);
  stage_tok[e0 * CAP + p] = t;
  stage_w  [e0 * CAP + p] = w0;
  if (e1 != e0) {
    p = atomicAdd(&cnt[e1], 1);
    stage_tok[e1 * CAP + p] = t;
    stage_w  [e1 * CAP + p] = w1;
  }
}

// ---------------------------------------------------------------------------
// Kernel 2: compact — pad per-expert segments to TILE_M, emit tile metadata.
// ---------------------------------------------------------------------------
__global__ void compact_kernel(const int* __restrict__ cnt,
                               const int* __restrict__ stage_tok,
                               const float* __restrict__ stage_w,
                               int* __restrict__ pair_tok,
                               float* __restrict__ pair_w,
                               int* __restrict__ tile_expert,
                               int* __restrict__ tile_base,
                               int* __restrict__ meta) {
  __shared__ int base[NUM_EXPERTS];
  __shared__ int padded[NUM_EXPERTS];
  if (threadIdx.x == 0) {
    int off = 0;
    for (int e = 0; e < NUM_EXPERTS; ++e) {
      base[e] = off;
      int c  = cnt[e];
      int pc = ((c + TILE_M - 1) / TILE_M) * TILE_M;
      padded[e] = pc;
      off += pc;
    }
    meta[0] = off / TILE_M;          // number of live m-tiles
    int ti = 0;
    for (int e = 0; e < NUM_EXPERTS; ++e)
      for (int k = 0; k < padded[e] / TILE_M; ++k) {
        tile_expert[ti] = e;
        tile_base[ti]   = base[e] + k * TILE_M;
        ++ti;
      }
  }
  __syncthreads();
  for (int e = 0; e < NUM_EXPERTS; ++e) {
    int c = cnt[e], pc = padded[e], b = base[e];
    for (int i = threadIdx.x; i < pc; i += blockDim.x) {
      if (i < c) { pair_tok[b + i] = stage_tok[e * CAP + i];
                   pair_w  [b + i] = stage_w  [e * CAP + i]; }
      else       { pair_tok[b + i] = -1; pair_w[b + i] = 0.0f; }
    }
  }
}

// ---------------------------------------------------------------------------
// Kernel 3: gate/up GEMM + SwiGLU.  Block tile: 256 (pairs) x 64 (I cols),
// K over H in steps of 32, double-buffered LDS.  8 waves as 4x2, each wave
// computes 64x32 for gate AND up (shared A fragments).
// ---------------------------------------------------------------------------
__global__ void __launch_bounds__(256)
gateup_kernel(const float* __restrict__ hidden,
              const float* __restrict__ gate_w,
              const float* __restrict__ up_w,
              const int* __restrict__ pair_tok,
              const int* __restrict__ tile_expert,
              const int* __restrict__ tile_base,
              const int* __restrict__ meta,
              u16* __restrict__ act) {
  if ((int)blockIdx.y >= meta[0]) return;
  const int e    = tile_expert[blockIdx.y];
  const int base = tile_base[blockIdx.y];
  const int n0   = blockIdx.x * 64;

  __shared__ u16 As[2][TILE_M][LDK];   // 256 x 32 bf16, K-contiguous
  __shared__ u16 BgT[2][64][LDK];      // 64(n) x 32(k), K-contiguous
  __shared__ u16 BuT[2][64][LDK];
  __shared__ int row_tok[TILE_M];

  const int tid  = threadIdx.x;
  const int lane = tid & 31;
  const int wave = tid >> 5;
  const int wm   = wave & 3;       // 4 row groups of 64
  const int wn   = wave >> 2;      // 2 col groups of 32
  const int h    = lane >> 4;
  const int l16  = lane & 15;

  for (int i = tid; i < TILE_M; i += 256) {
    int t = pair_tok[base + i];
    row_tok[i] = (t < 0) ? 0 : t;  // padded rows read token 0; weight is 0
  }
  __syncthreads();

  const float* gw = gate_w + (size_t)e * H_ * I_;
  const float* uw = up_w   + (size_t)e * H_ * I_;

  auto stage = [&](int buf, int k0) {
    // A: 256x32 f32 -> bf16, one ds_store_b128 per 8-element run
#pragma unroll
    for (int i = 0; i < 4; ++i) {
      int linear = i * 256 + tid;          // 0..1023
      int r  = linear >> 2;                // 0..255
      int c8 = (linear & 3) * 8;           // 0,8,16,24
      const float* src = &hidden[(size_t)row_tok[r] * H_ + k0 + c8];
      float4 f0 = *(const float4*)(src);
      float4 f1 = *(const float4*)(src + 4);
      *(uint4*)&As[buf][r][c8] = pk8(f0, f1);
    }
    // B (gate & up): 32(k) x 64(n) f32, transposed store as K-pairs (b32)
#pragma unroll
    for (int i = 0; i < 2; ++i) {
      int q  = i * 256 + tid;              // 0..511 quads of 2k x 2n
      int kq = q >> 5;                     // 0..15
      int nq = q & 31;                     // 0..31
      int k  = kq * 2, n = nq * 2;
      size_t g0 = (size_t)(k0 + k) * I_ + n0 + n;
      float2 ga = *(const float2*)&gw[g0];
      float2 gb = *(const float2*)&gw[g0 + I_];
      float2 ua = *(const float2*)&uw[g0];
      float2 ub = *(const float2*)&uw[g0 + I_];
      *(u32*)&BgT[buf][n + 0][k] = pk2(ga.x, gb.x);
      *(u32*)&BgT[buf][n + 1][k] = pk2(ga.y, gb.y);
      *(u32*)&BuT[buf][n + 0][k] = pk2(ua.x, ub.x);
      *(u32*)&BuT[buf][n + 1][k] = pk2(ua.y, ub.y);
    }
  };

  v8f accg[4][2], accu[4][2];
#pragma unroll
  for (int i = 0; i < 4; ++i)
#pragma unroll
    for (int j = 0; j < 2; ++j)
#pragma unroll
      for (int q = 0; q < 8; ++q) { accg[i][j][q] = 0.0f; accu[i][j][q] = 0.0f; }

  stage(0, 0);
  __syncthreads();

  const int NK = H_ / 32;
  for (int kt = 0; kt < NK; ++kt) {
    const int cur = kt & 1;
    if (kt + 1 < NK) stage(cur ^ 1, (kt + 1) * 32);

    Frag a[4];
#pragma unroll
    for (int mf = 0; mf < 4; ++mf) {
      int row = wm * 64 + mf * 16 + l16;
      a[mf].q[0] = *(const uint4*)&As[cur][row][8 * h];
      a[mf].q[1] = *(const uint4*)&As[cur][row][16 + 8 * h];
    }
#pragma unroll
    for (int nf = 0; nf < 2; ++nf) {
      int col = wn * 32 + nf * 16 + l16;
      Frag bg, bu;
      bg.q[0] = *(const uint4*)&BgT[cur][col][8 * h];
      bg.q[1] = *(const uint4*)&BgT[cur][col][16 + 8 * h];
      bu.q[0] = *(const uint4*)&BuT[cur][col][8 * h];
      bu.q[1] = *(const uint4*)&BuT[cur][col][16 + 8 * h];
#pragma unroll
      for (int mf = 0; mf < 4; ++mf) {
        accg[mf][nf] = __builtin_amdgcn_wmma_f32_16x16x32_bf16(
            false, a[mf].v, false, bg.v, (short)0, accg[mf][nf], false, false);
        accu[mf][nf] = __builtin_amdgcn_wmma_f32_16x16x32_bf16(
            false, a[mf].v, false, bu.v, (short)0, accu[mf][nf], false, false);
      }
    }
    __syncthreads();
  }

  // SwiGLU epilogue: act = silu(g) * u, stored bf16 pair-row major.
#pragma unroll
  for (int mf = 0; mf < 4; ++mf)
#pragma unroll
    for (int nf = 0; nf < 2; ++nf)
#pragma unroll
      for (int r = 0; r < 8; ++r) {
        int m = wm * 64 + mf * 16 + r + 8 * h;
        int n = wn * 32 + nf * 16 + l16;
        float g = accg[mf][nf][r];
        float u = accu[mf][nf][r];
        float s = g / (1.0f + __expf(-g));
        act[(size_t)(base + m) * I_ + n0 + n] = f2bf(s * u);
      }
}

// ---------------------------------------------------------------------------
// Kernel 4: down GEMM + weighted atomic scatter-add into d_out.
// Block tile: 256 (pairs) x 64 (H cols), K over I in steps of 32.
// ---------------------------------------------------------------------------
__global__ void __launch_bounds__(256)
down_kernel(const u16* __restrict__ act,
            const float* __restrict__ down_w,
            const int* __restrict__ pair_tok,
            const float* __restrict__ pair_w,
            const int* __restrict__ tile_expert,
            const int* __restrict__ tile_base,
            const int* __restrict__ meta,
            float* __restrict__ out) {
  if ((int)blockIdx.y >= meta[0]) return;
  const int e    = tile_expert[blockIdx.y];
  const int base = tile_base[blockIdx.y];
  const int n0   = blockIdx.x * 64;

  __shared__ u16   As[2][TILE_M][LDK];
  __shared__ u16   BdT[2][64][LDK];
  __shared__ int   row_tok[TILE_M];
  __shared__ float row_w[TILE_M];

  const int tid  = threadIdx.x;
  const int lane = tid & 31;
  const int wave = tid >> 5;
  const int wm   = wave & 3;
  const int wn   = wave >> 2;
  const int h    = lane >> 4;
  const int l16  = lane & 15;

  for (int i = tid; i < TILE_M; i += 256) {
    row_tok[i] = pair_tok[base + i];
    row_w[i]   = pair_w[base + i];
  }
  __syncthreads();

  const float* dw = down_w + (size_t)e * I_ * H_;

  auto stage = [&](int buf, int k0) {
    // A: bf16 activations, straight 16B copies
#pragma unroll
    for (int i = 0; i < 4; ++i) {
      int linear = i * 256 + tid;
      int r  = linear >> 2;
      int c8 = (linear & 3) * 8;
      *(uint4*)&As[buf][r][c8] =
          *(const uint4*)&act[(size_t)(base + r) * I_ + k0 + c8];
    }
    // B: 32(k) x 64(n) f32 down_w, transposed store as K-pairs
#pragma unroll
    for (int i = 0; i < 2; ++i) {
      int q  = i * 256 + tid;
      int kq = q >> 5;
      int nq = q & 31;
      int k  = kq * 2, n = nq * 2;
      size_t g0 = (size_t)(k0 + k) * H_ + n0 + n;
      float2 da = *(const float2*)&dw[g0];
      float2 db = *(const float2*)&dw[g0 + H_];
      *(u32*)&BdT[buf][n + 0][k] = pk2(da.x, db.x);
      *(u32*)&BdT[buf][n + 1][k] = pk2(da.y, db.y);
    }
  };

  v8f acc[4][2];
#pragma unroll
  for (int i = 0; i < 4; ++i)
#pragma unroll
    for (int j = 0; j < 2; ++j)
#pragma unroll
      for (int q = 0; q < 8; ++q) acc[i][j][q] = 0.0f;

  stage(0, 0);
  __syncthreads();

  const int NK = I_ / 32;
  for (int kt = 0; kt < NK; ++kt) {
    const int cur = kt & 1;
    if (kt + 1 < NK) stage(cur ^ 1, (kt + 1) * 32);

    Frag a[4];
#pragma unroll
    for (int mf = 0; mf < 4; ++mf) {
      int row = wm * 64 + mf * 16 + l16;
      a[mf].q[0] = *(const uint4*)&As[cur][row][8 * h];
      a[mf].q[1] = *(const uint4*)&As[cur][row][16 + 8 * h];
    }
#pragma unroll
    for (int nf = 0; nf < 2; ++nf) {
      int col = wn * 32 + nf * 16 + l16;
      Frag bd;
      bd.q[0] = *(const uint4*)&BdT[cur][col][8 * h];
      bd.q[1] = *(const uint4*)&BdT[cur][col][16 + 8 * h];
#pragma unroll
      for (int mf = 0; mf < 4; ++mf)
        acc[mf][nf] = __builtin_amdgcn_wmma_f32_16x16x32_bf16(
            false, a[mf].v, false, bd.v, (short)0, acc[mf][nf], false, false);
    }
    __syncthreads();
  }

  // weighted scatter-add: each output element receives at most TOPK adds
#pragma unroll
  for (int mf = 0; mf < 4; ++mf)
#pragma unroll
    for (int nf = 0; nf < 2; ++nf)
#pragma unroll
      for (int r = 0; r < 8; ++r) {
        int m = wm * 64 + mf * 16 + r + 8 * h;
        int t = row_tok[m];
        if (t >= 0) {
          int n = wn * 32 + nf * 16 + l16;
          atomicAdd(&out[(size_t)t * H_ + n0 + n], row_w[m] * acc[mf][nf][r]);
        }
      }
}

// ---------------------------------------------------------------------------
// Host-side launch (graph-capture safe: only async ops on `stream`).
// ---------------------------------------------------------------------------
extern "C" void kernel_launch(void* const* d_in, const int* in_sizes, int n_in,
                              void* d_out, int out_size, void* d_ws, size_t ws_size,
                              hipStream_t stream) {
  (void)in_sizes; (void)n_in; (void)ws_size;
  const int*   sel    = (const int*)  d_in[0];
  const float* hidden = (const float*)d_in[1];
  const float* rw     = (const float*)d_in[2];
  const float* gate_w = (const float*)d_in[3];
  const float* up_w   = (const float*)d_in[4];
  const float* down_w = (const float*)d_in[5];
  float*       out    = (float*)d_out;

  char* w = (char*)d_ws;
  const size_t OFF_CNT = 0;
  const size_t OFF_STT = 256;
  const size_t OFF_STW = OFF_STT + (size_t)NUM_EXPERTS * CAP * sizeof(int);
  const size_t OFF_PT  = OFF_STW + (size_t)NUM_EXPERTS * CAP * sizeof(float);
  const size_t OFF_PW  = OFF_PT  + (size_t)MAXSLOTS * sizeof(int);
  const size_t OFF_TE  = OFF_PW  + (size_t)MAXSLOTS * sizeof(float);
  const size_t OFF_TB  = OFF_TE  + 1024;
  const size_t OFF_ACT = OFF_TB  + 1024;

  int*   cnt         = (int*)  (w + OFF_CNT);
  int*   meta        = cnt + 16;
  int*   stage_tok   = (int*)  (w + OFF_STT);
  float* stage_w     = (float*)(w + OFF_STW);
  int*   pair_tok    = (int*)  (w + OFF_PT);
  float* pair_w      = (float*)(w + OFF_PW);
  int*   tile_expert = (int*)  (w + OFF_TE);
  int*   tile_base   = (int*)  (w + OFF_TB);
  u16*   act         = (u16*)  (w + OFF_ACT);

  hipMemsetAsync(cnt, 0, 256, stream);
  hipMemsetAsync(out, 0, (size_t)out_size * sizeof(float), stream);

  route_kernel<<<(NTOK + 255) / 256, 256, 0, stream>>>(sel, rw, cnt, stage_tok, stage_w);
  compact_kernel<<<1, 256, 0, stream>>>(cnt, stage_tok, stage_w,
                                        pair_tok, pair_w, tile_expert, tile_base, meta);
  gateup_kernel<<<dim3(I_ / 64, MAXTILES), 256, 0, stream>>>(
      hidden, gate_w, up_w, pair_tok, tile_expert, tile_base, meta, act);
  down_kernel<<<dim3(H_ / 64, MAXTILES), 256, 0, stream>>>(
      act, down_w, pair_tok, pair_w, tile_expert, tile_base, meta, out);
}
